// AttnPool_57294863729237
// MI455X (gfx1250) — matrix operand order repaired and checked by reading, not compiled, not run on hardware
//
#include <hip/hip_runtime.h>
#include <hip/hip_bf16.h>

// Problem constants (from reference): H=4, D=128, L1=L2=512, B=64
#define H_ 4
#define D_ 128
#define L_ 512
#define B_ 64

// Output layout (flat, in return order): r1f[64*128], r2f[64*128],
// a1[64*4*512], a2[64*4*512], adist[64*4]
#define OUT_R1F   0
#define OUT_R2F   8192
#define OUT_A1    16384
#define OUT_A2    147456
#define OUT_ADIST 278528

typedef __bf16 v16bf __attribute__((ext_vector_type(16)));
typedef __bf16 v8bf  __attribute__((ext_vector_type(8)));
typedef float  v8f   __attribute__((ext_vector_type(8)));

union V16U { v16bf v; v8bf h[2]; };

__device__ __forceinline__ float fast_tanh(float x) {
    // 1 - 2/(e^{2x}+1): saturates cleanly to +/-1 without inf/inf NaN
    float e = __expf(2.0f * x);
    return 1.0f - 2.0f / (e + 1.0f);
}
// order-preserving float -> u32 key for atomicMax-based float max
__device__ __forceinline__ unsigned f2key(float f) {
    unsigned u = __float_as_uint(f);
    return (u & 0x80000000u) ? ~u : (u | 0x80000000u);
}
__device__ __forceinline__ float key2f(unsigned k) {
    unsigned u = (k & 0x80000000u) ? (k & 0x7fffffffu) : ~k;
    return __uint_as_float(u);
}

// ---------------------------------------------------------------- prep ------
__global__ void prep_kernel(const float* __restrict__ x1f, const float* __restrict__ x2f,
                            const float* __restrict__ wuf,
                            const int* __restrict__ raw1, const int* __restrict__ raw2,
                            __bf16* __restrict__ x1b, __bf16* __restrict__ x2b,
                            __bf16* __restrict__ wub,
                            float* __restrict__ pen1, float* __restrict__ pen2,
                            unsigned* __restrict__ s2key) {
    const int stride = gridDim.x * blockDim.x;
    const int tid = blockIdx.x * blockDim.x + threadIdx.x;
    const int NB = L_ * B_ * D_;
    for (int i = tid; i < NB; i += stride) { x1b[i] = (__bf16)x1f[i]; x2b[i] = (__bf16)x2f[i]; }
    for (int i = tid; i < H_ * D_ * D_; i += stride) wub[i] = (__bf16)wuf[i];
    for (int i = tid; i < L_ * B_; i += stride) {
        pen1[i] = (raw1[i] == 0) ? -10000.0f : 0.0f;
        pen2[i] = (raw2[i] == 0) ? -10000.0f : 0.0f;
    }
    for (int i = tid; i < B_ * H_ * L_; i += stride) s2key[i] = 0u;  // ~= -inf key
}

// -------------------------------------------------- kA: Y = X2 * W_U[h]^T ---
// Rows = (m*B+b) flat (32768), N = d (128), K = e (128). bf16 WMMA, f32 acc.
// W_U[h] (32KB bf16) is shared by all waves and lives comfortably in L1/L2.
__global__ __launch_bounds__(256) void kA(const __bf16* __restrict__ x2b,
                                          const __bf16* __restrict__ wub,
                                          __bf16* __restrict__ ybf) {
    const int h      = blockIdx.x >> 8;
    const int rowblk = blockIdx.x & 255;
    const int w    = threadIdx.x >> 5;
    const int lane = threadIdx.x & 31;
    const int half = lane >> 4;
    const int mi   = lane & 15;
    const int row0 = rowblk * 128 + w * 16;

    // A fragments: 16 rows x K=128 (4 tiles of 16x32, contiguous k = 16*half + s per lane)
    const __bf16* arow = x2b + (size_t)(row0 + mi) * D_;
    v16bf a[4];
#pragma unroll
    for (int kt = 0; kt < 4; ++kt) {
        V16U u;
        const __bf16* p = arow + kt * 32 + half * 16;
        u.h[0] = *(const v8bf*)(p);
        u.h[1] = *(const v8bf*)(p + 8);
        a[kt] = u.v;
    }
    const __bf16* wh = wub + (size_t)h * D_ * D_;
    __bf16* yh = ybf + (size_t)h * L_ * B_ * D_;
    for (int nt = 0; nt < 8; ++nt) {
        const __bf16* brow = wh + (size_t)(nt * 16 + mi) * D_;  // B^T[n,k]=W_U[h][d=n,e=k]
        v8f c = {};
#pragma unroll
        for (int kt = 0; kt < 4; ++kt) {
            v16bf bt = *(const v16bf*)(brow + kt * 32 + half * 16);
            c = __builtin_amdgcn_wmma_f32_16x16x32_bf16(false, a[kt], false, bt,
                                                        (short)0, c, false, false);
        }
        const int col = nt * 16 + mi;
#pragma unroll
        for (int r = 0; r < 8; ++r)
            yh[(size_t)(row0 + half * 8 + r) * D_ + col] = (__bf16)c[r];
    }
}

// ---- kB: G[l,m]=tanh(x1 . Y), fused masked row/col max; G never stored ----
// grid: b(64) x h(4) x lchunk(4); block 256 = 8 waves; each wave owns 16 l-rows.
// The 16x128 Y tile for each m-step is shared by all 8 waves -> stage it once
// per block into LDS (double-buffered, 8KB) instead of 8x redundant L2 reads.
__global__ __launch_bounds__(256) void kB(const __bf16* __restrict__ x1b,
                                          const __bf16* __restrict__ ybf,
                                          const float* __restrict__ pen1,
                                          const float* __restrict__ pen2,
                                          float* __restrict__ s1,
                                          unsigned* __restrict__ s2key) {
    __shared__ __attribute__((aligned(32))) __bf16 sB[2][16 * D_];  // 2 x 4KB

    const int b  = blockIdx.x >> 4;
    const int h  = (blockIdx.x >> 2) & 3;
    const int lc = blockIdx.x & 3;
    const int w    = threadIdx.x >> 5;
    const int lane = threadIdx.x & 31;
    const int half = lane >> 4;
    const int mi   = lane & 15;
    const int l0 = lc * 128 + w * 16;

    const __bf16* ybase = ybf + (size_t)h * L_ * B_ * D_ + (size_t)b * D_;

    // cooperative tile stage: 256 threads x 16B = 4KB (16 rows x 256B)
    const int sr = threadIdx.x >> 4;   // row within tile 0..15
    const int sc = threadIdx.x & 15;   // 16B chunk within row 0..15
    auto stage = [&](int buf, int mt) {
        const v8bf* g = (const v8bf*)(ybase + (size_t)(mt * 16 + sr) * (B_ * D_) + sc * 8);
        *(v8bf*)(&sB[buf][sr * D_ + sc * 8]) = *g;
        if (mt + 1 < 32)
            __builtin_prefetch(ybase + (size_t)((mt + 1) * 16 + sr) * (B_ * D_) + sc * 8, 0, 1);
    };

    // A fragments: 16 rows of x1 (b fixed), K=D=128, held in registers
    const __bf16* arow = x1b + (size_t)((l0 + mi) * B_ + b) * D_;
    v16bf a[4];
#pragma unroll
    for (int kt = 0; kt < 4; ++kt) {
        V16U u;
        const __bf16* p = arow + kt * 32 + half * 16;
        u.h[0] = *(const v8bf*)(p);
        u.h[1] = *(const v8bf*)(p + 8);
        a[kt] = u.v;
    }
    float pl[8];
#pragma unroll
    for (int r = 0; r < 8; ++r) pl[r] = pen1[(l0 + half * 8 + r) * B_ + b];
    float rm[8];
#pragma unroll
    for (int r = 0; r < 8; ++r) rm[r] = -3.0e38f;

    float*    s1p = s1    + (size_t)(b * H_ + h) * L_;
    unsigned* s2p = s2key + (size_t)(b * H_ + h) * L_;

    stage(0, 0);
    __syncthreads();
    for (int mt = 0; mt < 32; ++mt) {
        const int nb = mt & 1;
        if (mt + 1 < 32) stage(1 - nb, mt + 1);  // fill next while computing

        v8f c = {};
#pragma unroll
        for (int kt = 0; kt < 4; ++kt) {
            v16bf bt = *(const v16bf*)(&sB[nb][mi * D_ + kt * 32 + half * 16]);
            c = __builtin_amdgcn_wmma_f32_16x16x32_bf16(false, a[kt], false, bt,
                                                        (short)0, c, false, false);
        }
        const float p2 = pen2[(mt * 16 + mi) * B_ + b];  // -10000 if m padded
        float cmax = -3.0e38f;
#pragma unroll
        for (int r = 0; r < 8; ++r) {
            float g = fast_tanh(c[r]);
            rm[r] = fmaxf(rm[r], g + p2);        // s1: max over m (per l row)
            cmax  = fmaxf(cmax,  g + pl[r]);     // s2: max over l (per m col)
        }
        cmax = fmaxf(cmax, __shfl_xor(cmax, 16, 32));  // combine M halves
        atomicMax(&s2p[mt * 16 + mi], f2key(cmax));    // cross-block col max
        __syncthreads();  // reads of buf nb done; stage writes visible
    }
    // s1 row-max: reduce across the 16 lanes of each half-group
#pragma unroll
    for (int off = 1; off <= 8; off <<= 1) {
#pragma unroll
        for (int r = 0; r < 8; ++r) rm[r] = fmaxf(rm[r], __shfl_xor(rm[r], off, 32));
    }
    if (mi == 0) {
#pragma unroll
        for (int r = 0; r < 8; ++r) s1p[l0 + half * 8 + r] = rm[r];
    }
}

// --------------------------------- kC: softmax over L for s1->a1, s2->a2 ---
__global__ __launch_bounds__(512) void kC(const float* __restrict__ s1,
                                          const unsigned* __restrict__ s2key,
                                          float* __restrict__ out) {
    __shared__ float red[512];
    const int which = blockIdx.x >> 8;   // 0 -> a1, 1 -> a2
    const int bh = blockIdx.x & 255;
    const int t = threadIdx.x;
    float v = (which == 0) ? s1[(size_t)bh * L_ + t] : key2f(s2key[(size_t)bh * L_ + t]);
    red[t] = v; __syncthreads();
    for (int s = 256; s > 0; s >>= 1) { if (t < s) red[t] = fmaxf(red[t], red[t + s]); __syncthreads(); }
    float m = red[0]; __syncthreads();
    float e = __expf(v - m);
    red[t] = e; __syncthreads();
    for (int s = 256; s > 0; s >>= 1) { if (t < s) red[t] += red[t + s]; __syncthreads(); }
    float a = e / red[0];
    float* dst = out + ((which == 0) ? OUT_A1 : OUT_A2);
    dst[(size_t)bh * L_ + t] = a;
}

// ------------------- kD: r1[b,h,:] = sum_l a1[b,h,l]*x1[l,b,:] (f32) -------
__global__ __launch_bounds__(128) void kD(const float* __restrict__ x1f,
                                          const float* __restrict__ x2f,
                                          const float* __restrict__ out,
                                          float* __restrict__ r1, float* __restrict__ r2) {
    __shared__ float sa[512];
    const int b = blockIdx.x >> 2;
    const int h = blockIdx.x & 3;
    const int t = threadIdx.x;
    const int bh = b * H_ + h;
    const float* a1 = out + OUT_A1 + (size_t)bh * L_;
    const float* a2 = out + OUT_A2 + (size_t)bh * L_;
    for (int i = t; i < L_; i += 128) sa[i] = a1[i];
    __syncthreads();
    float acc = 0.0f;
    for (int l = 0; l < L_; ++l) acc = fmaf(sa[l], x1f[((size_t)l * B_ + b) * D_ + t], acc);
    r1[(size_t)bh * D_ + t] = acc;
    __syncthreads();
    for (int i = t; i < L_; i += 128) sa[i] = a2[i];
    __syncthreads();
    acc = 0.0f;
    for (int l = 0; l < L_; ++l) acc = fmaf(sa[l], x2f[((size_t)l * B_ + b) * D_ + t], acc);
    r2[(size_t)bh * D_ + t] = acc;
}

// ------ kE: IPM, hop softmax (adist), final combine r1f/r2f; one block/b ---
__global__ __launch_bounds__(128) void kE(const float* __restrict__ r1,
                                          const float* __restrict__ r2,
                                          const float* __restrict__ wipm,
                                          float* __restrict__ out) {
    __shared__ float sr1[512], sr2[512], sred[128], st[4], sad[4];
    const int b = blockIdx.x;
    const int t = threadIdx.x;
    for (int i = t; i < 512; i += 128) {
        sr1[i] = r1[(size_t)b * 512 + i];
        sr2[i] = r2[(size_t)b * 512 + i];
    }
    __syncthreads();
    float prod[4];
#pragma unroll
    for (int h = 0; h < 4; ++h) {
        float acc = 0.0f;
        for (int e = 0; e < D_; ++e) acc = fmaf(sr2[h * D_ + e], wipm[t * D_ + e], acc);
        prod[h] = sr1[h * D_ + t] * acc;   // r1 * IPM(r2), summed over d next
    }
    for (int h = 0; h < 4; ++h) {
        sred[t] = prod[h]; __syncthreads();
        for (int s = 64; s > 0; s >>= 1) { if (t < s) sred[t] += sred[t + s]; __syncthreads(); }
        if (t == 0) st[h] = sred[0];
        __syncthreads();
    }
    if (t == 0) {
        float th[4], mx = -3.0e38f;
        for (int h = 0; h < 4; ++h) { th[h] = fast_tanh(st[h]); mx = fmaxf(mx, th[h]); }
        float s = 0.0f;
        for (int h = 0; h < 4; ++h) { th[h] = __expf(th[h] - mx); s += th[h]; }
        for (int h = 0; h < 4; ++h) sad[h] = th[h] / s;
    }
    __syncthreads();
    float f1 = 0.0f, f2 = 0.0f;
#pragma unroll
    for (int h = 0; h < 4; ++h) { f1 += sad[h] * sr1[h * D_ + t]; f2 += sad[h] * sr2[h * D_ + t]; }
    out[OUT_R1F + (size_t)b * D_ + t] = f1;
    out[OUT_R2F + (size_t)b * D_ + t] = f2;
    if (t < 4) out[OUT_ADIST + b * 4 + t] = sad[t];
}

// ---------------------------------------------------------------------------
extern "C" void kernel_launch(void* const* d_in, const int* in_sizes, int n_in,
                              void* d_out, int out_size, void* d_ws, size_t ws_size,
                              hipStream_t stream) {
    (void)in_sizes; (void)n_in; (void)out_size; (void)ws_size;
    const float* x1f  = (const float*)d_in[0];   // [L1,B,D]
    const float* x2f  = (const float*)d_in[1];   // [L2,B,D]
    const int*   raw1 = (const int*)d_in[2];     // [L1,B]
    const int*   raw2 = (const int*)d_in[3];     // [L2,B]
    const float* wuf  = (const float*)d_in[4];   // [H,D,D]
    const float* wipm = (const float*)d_in[5];   // [D,D]
    float* out = (float*)d_out;

    // workspace carve (~50 MB total)
    char* ws = (char*)d_ws;
    size_t off = 0;
    auto carve = [&](size_t bytes) -> char* {
        char* p = ws + off;
        off += (bytes + 255) & ~(size_t)255;
        return p;
    };
    __bf16*   x1b  = (__bf16*)carve((size_t)L_ * B_ * D_ * 2);       // 8 MB
    __bf16*   x2b  = (__bf16*)carve((size_t)L_ * B_ * D_ * 2);       // 8 MB
    __bf16*   wub  = (__bf16*)carve((size_t)H_ * D_ * D_ * 2);       // 128 KB
    __bf16*   ybf  = (__bf16*)carve((size_t)H_ * L_ * B_ * D_ * 2);  // 32 MB
    float*    pen1 = (float*)carve((size_t)L_ * B_ * 4);
    float*    pen2 = (float*)carve((size_t)L_ * B_ * 4);
    float*    s1   = (float*)carve((size_t)B_ * H_ * L_ * 4);
    unsigned* s2k  = (unsigned*)carve((size_t)B_ * H_ * L_ * 4);
    float*    r1   = (float*)carve((size_t)B_ * H_ * D_ * 4);
    float*    r2   = (float*)carve((size_t)B_ * H_ * D_ * 4);

    prep_kernel<<<2048, 256, 0, stream>>>(x1f, x2f, wuf, raw1, raw2,
                                          x1b, x2b, wub, pen1, pen2, s2k);
    kA<<<1024, 256, 0, stream>>>(x2b, wub, ybf);
    kB<<<1024, 256, 0, stream>>>(x1b, ybf, pen1, pen2, s1, s2k);
    kC<<<512, 512, 0, stream>>>(s1, s2k, out);
    kD<<<256, 128, 0, stream>>>(x1f, x2f, out, r1, r2);
    kE<<<64, 128, 0, stream>>>(r1, r2, wipm, out);
}
